// MultiHeadAttention_7069516169536
// MI455X (gfx1250) — matrix operand order repaired and checked by reading, not compiled, not run on hardware
//
#include <hip/hip_runtime.h>

#define BATCH 8
#define SEQ   1024
#define EMB   1024
#define HEADS 16
#define HDIM  64
#define AWAVES 4   // waves per attention block

typedef __attribute__((ext_vector_type(16))) _Float16 v16h;
typedef __attribute__((ext_vector_type(8)))  _Float16 v8h;
typedef __attribute__((ext_vector_type(4)))  _Float16 v4h;
typedef __attribute__((ext_vector_type(8)))  float    v8f;
typedef __attribute__((ext_vector_type(4)))  float    v4f;
typedef unsigned int u32x4 __attribute__((ext_vector_type(4)));
typedef int          i32x4 __attribute__((ext_vector_type(4)));
typedef int          i32x8 __attribute__((ext_vector_type(8)));

#if __has_include(<hip/amd_detail/amd_gfx1250_TDM.h>)
#define TDM_6ARG 1
#else
#define TDM_6ARG 0
#endif

static __device__ __forceinline__ v8f wmma16(v16h a, v16h b, v8f c) {
  // D = A(16x32 f16) * B(32x16 f16) + C(16x16 f32)
  return __builtin_amdgcn_wmma_f32_16x16x32_f16(
      /*neg_a=*/false, a, /*neg_b=*/false, b,
      /*c_mod=*/(short)0, c, /*reuse_a=*/false, /*reuse_b=*/false);
}

// ---------------------------------------------------------------------------
// Tensor Data Mover: 2D tile (f16 elements) global -> LDS.
//   tile_d0 = contiguous elements per row, tile_d1 = rows,
//   row_stride = elements between row starts in memory.
// D# packed per CDNA5 ISA ch.8: group0 {count=1, lds_addr, global_addr,
// type=2}, group1 {data_size=1(2B), tensor_dim0/1 == tile dims, tile dims,
// tensor_dim0_stride}. Groups 2/3 zero (<=2D tensor).
// ---------------------------------------------------------------------------
static __device__ __forceinline__ void tdm_load_2d(const void* gptr,
                                                   unsigned lds_byte_off,
                                                   unsigned tile_d0,
                                                   unsigned tile_d1,
                                                   unsigned row_stride) {
  unsigned long long ga = (unsigned long long)gptr;
  u32x4 g0;
  g0[0] = 1u;                                        // count=1, user-mode D#
  g0[1] = lds_byte_off;                              // lds_addr (bytes)
  g0[2] = (unsigned)(ga & 0xffffffffu);              // global_addr[31:0]
  g0[3] = (unsigned)((ga >> 32) & 0x01ffffffu)       // global_addr[56:32]
          | (2u << 30);                              // type=2 ("image")
  i32x8 g1;
  g1[0] = 0x00010000;                                // data_size=1 -> 2 bytes
  g1[1] = (int)((tile_d0 & 0xffffu) << 16);          // tensor_dim0[15:0]
  g1[2] = (int)(((tile_d0 >> 16) & 0xffffu) |        // tensor_dim0[31:16]
                ((tile_d1 & 0xffffu) << 16));        // tensor_dim1[15:0]
  g1[3] = (int)(((tile_d1 >> 16) & 0xffffu) |        // tensor_dim1[31:16]
                ((tile_d0 & 0xffffu) << 16));        // tile_dim0
  g1[4] = (int)(tile_d1 & 0xffffu);                  // tile_dim1 (tile_dim2=0)
  g1[5] = (int)row_stride;                           // tensor_dim0_stride lo32
  g1[6] = 0;                                         // stride hi / dim1_stride
  g1[7] = 0;
  i32x4 g2 = {0, 0, 0, 0};
  i32x4 g3 = {0, 0, 0, 0};
#if TDM_6ARG
  i32x8 g4 = {0, 0, 0, 0, 0, 0, 0, 0};
  __builtin_amdgcn_tensor_load_to_lds(g0, g1, g2, g3, g4, 0);
#else
  __builtin_amdgcn_tensor_load_to_lds(g0, g1, g2, g3, 0);
#endif
}

// ---------------------------------------------------------------------------
// f32 -> f16 conversion (vectorized by 4)
// ---------------------------------------------------------------------------
__global__ void cvt_f32_to_f16(const float* __restrict__ in,
                               _Float16* __restrict__ out, int n4) {
  int i = blockIdx.x * blockDim.x + threadIdx.x;
  if (i < n4) {
    v4f v = *(const v4f*)(in + (size_t)i * 4);
    v4h h;
    h[0] = (_Float16)v[0]; h[1] = (_Float16)v[1];
    h[2] = (_Float16)v[2]; h[3] = (_Float16)v[3];
    *(v4h*)(out + (size_t)i * 4) = h;
  }
}

// ---------------------------------------------------------------------------
// Generic WMMA GEMM:  out[m][n] = sum_k A[m][k] * W[n][k] + bias[n]
// A: MxK f16 row-major, W: NxK f16 row-major (torch Linear weight layout).
// Block = 8 waves sharing one 16-row A strip (identical A fragment
// addresses -> WGP$ hits); wave w owns output tile (tm, blockIdx.y*8+w).
// mode 0: f16 row-major MxN        (Q, K projections)
// mode 1: f16 per-head transposed  (V projection -> Vt[(b,h,d,s)])
// mode 2: f32 row-major MxN        (final output projection)
// ---------------------------------------------------------------------------
__global__ void gemm_wmma_f16(const _Float16* __restrict__ A,
                              const _Float16* __restrict__ W,
                              const float* __restrict__ bias,
                              void* __restrict__ out,
                              int M, int N, int K, int mode) {
  const int wave = threadIdx.x >> 5;
  const int lane = threadIdx.x & 31;
  const int lr = lane & 15;   // row-within-tile for A, col for B
  const int ks = lane >> 4;   // K-half selector
  const int tm = blockIdx.x;
  const int tn = blockIdx.y * 8 + wave;

  const _Float16* ap = A + (size_t)(tm * 16 + lr) * K + ks * 16;
  const _Float16* wp = W + (size_t)(tn * 16 + lr) * K + ks * 16;

  v8f c = {};
  for (int k0 = 0; k0 < K; k0 += 32) {
    union { v16h v; v8h h[2]; } af;
    af.h[0] = *(const v8h*)(ap);        // K = ks*16 .. +7
    af.h[1] = *(const v8h*)(ap + 8);    // K = ks*16 + 8 .. +15
    v16h bf = *(const v16h*)(wp);       // K = ks*16 .. +15 for column lr
    __builtin_prefetch(ap + 32);        // global_prefetch_b8, next K tile
    __builtin_prefetch(wp + 32);
    c = wmma16(af.v, bf, c);
    ap += 32;
    wp += 32;
  }

  const int n = tn * 16 + lr;
  const float bv = bias[n];

  if (mode == 0) {
    _Float16* o = (_Float16*)out;
#pragma unroll
    for (int r = 0; r < 8; ++r) {
      int m = tm * 16 + r + 8 * ks;
      o[(size_t)m * N + n] = (_Float16)(c[r] + bv);
    }
  } else if (mode == 1) {
    // out col n = h*64 + d ; out row m = b*SEQ + s ; store Vt[(b,h,d,s)]
    _Float16* o = (_Float16*)out;
    const int hh = n >> 6, d = n & 63;
#pragma unroll
    for (int r = 0; r < 8; ++r) {
      int m = tm * 16 + r + 8 * ks;
      int b = m >> 10, s = m & 1023;
      o[(((size_t)b * HEADS + hh) * HDIM + d) * SEQ + s] = (_Float16)(c[r] + bv);
    }
  } else {
    float* o = (float*)out;
#pragma unroll
    for (int r = 0; r < 8; ++r) {
      int m = tm * 16 + r + 8 * ks;
      o[(size_t)m * N + n] = c[r] + bv;
    }
  }
}

// ---------------------------------------------------------------------------
// Attention core: one wave owns a 16-query tile for one (b,h).
// TDM stages the K head-slab (1024x64 f16, 128 KB) into LDS for QK^T, then
// (async, overlapped with softmax) restages the same LDS region with the
// transposed V head-slab for P@V. Scores/probs kept in LDS as f16 (32 KB
// per wave). Block = 4 waves (one per SIMD32 of the WGP);
// dynamic LDS = 4K + 4*32K + 128K = 260 KB (<= 320 KB WGP budget).
// ---------------------------------------------------------------------------
__global__ void attn_wmma(const _Float16* __restrict__ Qh,
                          const _Float16* __restrict__ Kh,
                          const _Float16* __restrict__ Vt,
                          const int* __restrict__ mask,
                          float* __restrict__ aw,
                          _Float16* __restrict__ att) {
  extern __shared__ char smem[];
  const int b = blockIdx.z, h = blockIdx.y;
  const int wave = threadIdx.x >> 5;
  const int lane = threadIdx.x & 31;
  const int lr = lane & 15, ks = lane >> 4;
  const int q0 = (blockIdx.x * AWAVES + wave) * 16;

  float*    maskAdd = (float*)smem;                                // 4 KB
  _Float16* sc   = (_Float16*)(smem + 4096 + wave * 32768);        // 16xSEQ f16
  _Float16* slab = (_Float16*)(smem + 4096 + AWAVES * 32768);      // 128 KB
  const unsigned slabOff = 4096u + AWAVES * 32768u;                // TDM addr

  // Kick off TDM for the K head-slab: 1024 rows x 64 f16, row stride EMB.
  if (wave == 0)
    tdm_load_2d(Kh + (size_t)(b * SEQ) * EMB + h * HDIM, slabOff,
                HDIM, SEQ, EMB);

  for (int i = threadIdx.x; i < SEQ; i += blockDim.x)
    maskAdd[i] = (mask[b * SEQ + i] == 0) ? -60000.0f : 0.0f;

  // --- Q fragments (16 rows x 64 of D, two K=32 chunks), from global ---
  union { v16h v; v8h h2[2]; } qa[2];
  {
    const _Float16* qrow = Qh + ((size_t)(b * SEQ + q0 + lr)) * EMB + h * HDIM;
#pragma unroll
    for (int j = 0; j < 2; ++j) {
      const _Float16* p = qrow + j * 32 + ks * 16;
      qa[j].h2[0] = *(const v8h*)(p);
      qa[j].h2[1] = *(const v8h*)(p + 8);
    }
  }

  if (wave == 0) __builtin_amdgcn_s_wait_tensorcnt(0);  // K slab resident
  __syncthreads();

  // --- scores = clip(QK^T / 8) + maskAdd, K tiles from the LDS slab ---
  for (int kt = 0; kt < SEQ / 16; ++kt) {
    const int n0 = kt * 16;
    const _Float16* kr = slab + (size_t)(n0 + lr) * HDIM + ks * 16;
    v16h b0 = *(const v16h*)(kr);        // d = ks*16 .. +15
    v16h b1 = *(const v16h*)(kr + 32);   // d = 32 + ks*16 .. +15
    v8f c = {};
    c = wmma16(qa[0].v, b0, c);
    c = wmma16(qa[1].v, b1, c);
    const int n = n0 + lr;
    const float madd = maskAdd[n];
#pragma unroll
    for (int r = 0; r < 8; ++r) {
      float s = c[r] * 0.125f;  // 1/sqrt(64)
      s = fminf(50.0f, fmaxf(-50.0f, s)) + madd;
      sc[(r + 8 * ks) * SEQ + n] = (_Float16)s;
    }
  }
  __syncthreads();  // all waves done with K slab + scores written

  // Restage slab with transposed V head-slab (64 rows x 1024 f16,
  // contiguous) -- DMA overlaps with softmax below.
  if (wave == 0)
    tdm_load_2d(Vt + (size_t)(b * HEADS + h) * HDIM * SEQ, slabOff,
                SEQ, HDIM, SEQ);

  // --- softmax: lane pair (lr, lr+16) owns row lr, 512 cols each ---
  {
    v8h* sr = (v8h*)(sc + lr * SEQ + ks * 512);
    float mx = -3.0e38f;
    for (int i = 0; i < 64; ++i) {
      v8h t = sr[i];
#pragma unroll
      for (int j = 0; j < 8; ++j) mx = fmaxf(mx, (float)t[j]);
    }
    mx = fmaxf(mx, __shfl_xor(mx, 16, 32));
    float sum = 0.0f;
    for (int i = 0; i < 64; ++i) {
      v8h t = sr[i], e;
#pragma unroll
      for (int j = 0; j < 8; ++j) {
        float ef = __expf((float)t[j] - mx);
        e[j] = (_Float16)ef;
        sum += ef;
      }
      sr[i] = e;
    }
    sum += __shfl_xor(sum, 16, 32);
    const float inv = 1.0f / (sum + 1e-10f);
    v4f* awp = (v4f*)(aw + ((size_t)(b * HEADS + h) * SEQ + q0 + lr) * SEQ +
                      ks * 512);
    for (int i = 0; i < 64; ++i) {
      v8h t = sr[i], p;
      v4f a0, a1;
#pragma unroll
      for (int j = 0; j < 4; ++j) {
        float p0 = (float)t[j] * inv;
        float p1 = (float)t[4 + j] * inv;
        p[j] = (_Float16)p0;
        p[4 + j] = (_Float16)p1;
        a0[j] = p0;
        a1[j] = p1;
      }
      sr[i] = p;          // normalized probs stay in LDS (f16) for P@V
      awp[2 * i] = a0;    // single mandatory HBM write of aw (f32)
      awp[2 * i + 1] = a1;
    }
  }

  if (wave == 0) __builtin_amdgcn_s_wait_tensorcnt(0);  // V slab resident
  __syncthreads();

  // --- attended = P(16 x SEQ f16, LDS) @ V (LDS slab is (d,s)) ---
  v8f acc[4] = {};
  for (int k0 = 0; k0 < SEQ; k0 += 32) {
    union { v16h v; v8h h2[2]; } pa;
    const _Float16* pr = sc + lr * SEQ + k0 + ks * 16;
    pa.h2[0] = *(const v8h*)(pr);
    pa.h2[1] = *(const v8h*)(pr + 8);
#pragma unroll
    for (int nt = 0; nt < 4; ++nt) {
      const int d = nt * 16 + lr;
      v16h bf = *(const v16h*)(slab + (size_t)d * SEQ + k0 + ks * 16);
      acc[nt] = wmma16(pa.v, bf, acc[nt]);
    }
  }
#pragma unroll
  for (int nt = 0; nt < 4; ++nt) {
#pragma unroll
    for (int r = 0; r < 8; ++r) {
      int m = q0 + r + 8 * ks;
      int d = nt * 16 + lr;
      att[((size_t)(b * SEQ + m)) * EMB + h * HDIM + d] = (_Float16)acc[nt][r];
    }
  }
}

// ---------------------------------------------------------------------------
extern "C" void kernel_launch(void* const* d_in, const int* in_sizes, int n_in,
                              void* d_out, int out_size, void* d_ws,
                              size_t ws_size, hipStream_t stream) {
  (void)in_sizes; (void)n_in; (void)out_size; (void)ws_size;

  const float* x    = (const float*)d_in[0];
  const int*   mask = (const int*)d_in[1];
  const float* Wq   = (const float*)d_in[2];
  const float* bq   = (const float*)d_in[3];
  const float* Wk   = (const float*)d_in[4];
  const float* bk   = (const float*)d_in[5];
  const float* Wv   = (const float*)d_in[6];
  const float* bv   = (const float*)d_in[7];
  const float* Wo   = (const float*)d_in[8];
  const float* bo   = (const float*)d_in[9];

  const size_t NTOK = (size_t)BATCH * SEQ;          // 8192
  const size_t NX   = NTOK * EMB;                   // 8M elems
  const size_t NW   = (size_t)EMB * EMB;            // 1M elems

  _Float16* Xh  = (_Float16*)d_ws;
  _Float16* Wqh = Xh  + NX;
  _Float16* Wkh = Wqh + NW;
  _Float16* Wvh = Wkh + NW;
  _Float16* Woh = Wvh + NW;
  _Float16* Qh  = Woh + NW;
  _Float16* Kh  = Qh  + NX;
  _Float16* Vt  = Kh  + NX;   // layout (B,H,D,S)
  _Float16* Att = Vt  + NX;   // layout (B,S,E), f16

  float* out = (float*)d_out;
  float* aw  = out + NX;      // (B,H,S,S) f32

  // f32 -> f16 conversions
  {
    int n4 = (int)(NX / 4);
    cvt_f32_to_f16<<<(n4 + 255) / 256, 256, 0, stream>>>(x, Xh, n4);
    n4 = (int)(NW / 4);
    cvt_f32_to_f16<<<(n4 + 255) / 256, 256, 0, stream>>>(Wq, Wqh, n4);
    cvt_f32_to_f16<<<(n4 + 255) / 256, 256, 0, stream>>>(Wk, Wkh, n4);
    cvt_f32_to_f16<<<(n4 + 255) / 256, 256, 0, stream>>>(Wv, Wvh, n4);
    cvt_f32_to_f16<<<(n4 + 255) / 256, 256, 0, stream>>>(Wo, Woh, n4);
  }

  // QKV projections: 8 waves/block share one A strip, cover 128 columns
  dim3 gg((unsigned)(NTOK / 16), EMB / 128);
  gemm_wmma_f16<<<gg, 256, 0, stream>>>(Xh, Wqh, bq, (void*)Qh,
                                        (int)NTOK, EMB, EMB, 0);
  gemm_wmma_f16<<<gg, 256, 0, stream>>>(Xh, Wkh, bk, (void*)Kh,
                                        (int)NTOK, EMB, EMB, 0);
  gemm_wmma_f16<<<gg, 256, 0, stream>>>(Xh, Wvh, bv, (void*)Vt,
                                        (int)NTOK, EMB, EMB, 1);

  // attention (TDM-staged K/V slabs, scores -> softmax -> aw + P@V)
  dim3 ga(SEQ / 16 / AWAVES, HEADS, BATCH);
  size_t shmem = 4096 + AWAVES * 32768 + 131072;  // 260 KB
  (void)hipFuncSetAttribute((const void*)attn_wmma,
                            hipFuncAttributeMaxDynamicSharedMemorySize,
                            (int)shmem);
  attn_wmma<<<ga, AWAVES * 32, shmem, stream>>>(Qh, Kh, Vt, mask, aw, Att);

  // output projection (f32 result)
  gemm_wmma_f16<<<gg, 256, 0, stream>>>(Att, Woh, bo, d_out,
                                        (int)NTOK, EMB, EMB, 2);
}
